// Robust_RPM_SVDHead_84464826843693
// MI455X (gfx1250) — matrix operand order repaired and checked by reading, not compiled, not run on hardware
//
#include <hip/hip_runtime.h>
#include <hip/hip_bf16.h>

typedef _Float16 f16;
typedef __attribute__((ext_vector_type(16))) _Float16 v16h;
typedef __attribute__((ext_vector_type(8)))  float    v8f;

#define B_SZ   8
#define D_DIM  256
#define N_PTS  2048
#define LA_ROWS 2049
#define LA_COLS 2064   // padded row stride (multiple of 16 floats)

// ---------------------------------------------------------------------------
__global__ __launch_bounds__(256) void zero_kernel(float* p, long long n) {
  long long i = (long long)blockIdx.x * blockDim.x + threadIdx.x;
  long long stride = (long long)gridDim.x * blockDim.x;
  for (; i < n; i += stride) p[i] = 0.0f;
}

// [B,256,2048] fp32 -> [B,2048,256] f16 (tiled transpose through LDS)
__global__ __launch_bounds__(256) void transpose_f16_kernel(
    const float* __restrict__ in, f16* __restrict__ out) {
  int b = blockIdx.z;
  int n0 = blockIdx.x * 32, d0 = blockIdx.y * 32;
  int tx = threadIdx.x, ty = threadIdx.y;             // 32 x 8
  __shared__ float tile[32][33];
  const float* ib = in + (size_t)b * D_DIM * N_PTS;
  f16* ob = out + (size_t)b * N_PTS * D_DIM;
  for (int i = 0; i < 4; ++i)
    tile[ty + i * 8][tx] = ib[(size_t)(d0 + ty + i * 8) * N_PTS + n0 + tx];
  __syncthreads();
  for (int i = 0; i < 4; ++i)
    ob[(size_t)(n0 + ty + i * 8) * D_DIM + d0 + tx] = (f16)tile[tx][ty + i * 8];
}

// squared norms per column, fp32 from original [B,256,2048]
__global__ __launch_bounds__(256) void norm_kernel(
    const float* __restrict__ in, float* __restrict__ out) {
  int b = blockIdx.y;
  int n = blockIdx.x * blockDim.x + threadIdx.x;
  const float* ib = in + (size_t)b * D_DIM * N_PTS;
  float s = 0.0f;
  for (int d = 0; d < D_DIM; ++d) {
    float x = ib[(size_t)d * N_PTS + n];
    s += x * x;
  }
  out[(size_t)b * N_PTS + n] = s;
}

// ---------------------------------------------------------------------------
// scores GEMM: la[b,n,m] = 0.2 - (xx[n] + yy[m] - 2*dot(x_n,y_m)) / 16
// 128x128 tile / block; 8 waves in a 4x2 grid, each wave computes 32x64
// (2 A-frags x 4 B-frags -> 8 accumulators) per 32-deep K-step using
// v_wmma_f32_16x16x32_f16. Double-buffered LDS; all fragments preloaded.
__global__ __launch_bounds__(256, 1) void gemm_scores_kernel(
    const f16* __restrict__ srcH, const f16* __restrict__ tgtH,
    const float* __restrict__ xx, const float* __restrict__ yy,
    float* __restrict__ la) {
  const int b  = blockIdx.z;
  const int n0 = blockIdx.y * 128;
  const int m0 = blockIdx.x * 128;
  const int tid  = threadIdx.x;
  const int lane = tid & 31;
  const int wv   = tid >> 5;
  const int wr   = wv >> 1;            // wave row 0..3 -> 32 rows each
  const int wc   = wv & 1;             // wave col 0..1 -> 64 cols each

  __shared__ __align__(16) f16 lA[2][128 * 40];  // 40-half stride: 16B-aligned rows
  __shared__ __align__(16) f16 lB[2][128 * 40];

  const f16* As = srcH + (size_t)b * N_PTS * D_DIM;
  const f16* Bs = tgtH + (size_t)b * N_PTS * D_DIM;

  union frag { v16h v; uint4 u[2]; };
  union { v8f v; float f[8]; } acc[2][4];
  for (int i = 0; i < 2; ++i)
    for (int j = 0; j < 4; ++j)
      for (int r = 0; r < 8; ++r) acc[i][j].f[r] = 0.0f;

  const int aM  = lane & 15;           // fragment row/col within 16
  const int kb  = (lane >> 4) * 8;     // A-frag K base  (ISA 16-bit A 16x32 layout)
  const int kbB = (lane >> 4) * 8;     // B-frag K base  (same K order as A)

  const int sr = tid >> 2;             // staging: row 0..63 (+64 second pass)
  const int sq = tid & 3;              // staging: 8-half chunk 0..3

  // stage k-slice [k0,k0+32) into buffer `buf`
  auto stage = [&](int k0, int buf) {
    f16* dA = lA[buf];
    f16* dB = lB[buf];
    int r = sr, q = sq;
    *(uint4*)&dA[r * 40 + q * 8] = *(const uint4*)&As[(size_t)(n0 + r) * D_DIM + k0 + q * 8];
    *(uint4*)&dB[r * 40 + q * 8] = *(const uint4*)&Bs[(size_t)(m0 + r) * D_DIM + k0 + q * 8];
    r = sr + 64;
    *(uint4*)&dA[r * 40 + q * 8] = *(const uint4*)&As[(size_t)(n0 + r) * D_DIM + k0 + q * 8];
    *(uint4*)&dB[r * 40 + q * 8] = *(const uint4*)&Bs[(size_t)(m0 + r) * D_DIM + k0 + q * 8];
  };

  stage(0, 0);
  __syncthreads();

  for (int ks = 0; ks < 8; ++ks) {
    const int cur = ks & 1;
    const f16* cA = lA[cur];
    const f16* cB = lB[cur];

    // load this step's fragments FIRST so staging stores don't gate the WMMAs
    frag af[2];
    for (int i = 0; i < 2; ++i) {
      const f16* pa = &cA[(wr * 32 + i * 16 + aM) * 40 + kb];
      af[i].u[0] = *(const uint4*)pa;          // K = kb .. kb+7
      af[i].u[1] = *(const uint4*)(pa + 16);   // K = kb+16 .. kb+23
    }
    frag bf[4];
    for (int j = 0; j < 4; ++j) {
      const f16* pb = &cB[(wc * 64 + j * 16 + aM) * 40 + kbB];
      bf[j].u[0] = *(const uint4*)pb;          // K = kbB .. kbB+7
      bf[j].u[1] = *(const uint4*)(pb + 16);   // K = kbB+16 .. kbB+23
    }

    // prefetch next k-slice into the other buffer while WMMAs run
    if (ks < 7) stage((ks + 1) * 32, cur ^ 1);

    for (int i = 0; i < 2; ++i)
      for (int j = 0; j < 4; ++j)
        acc[i][j].v = __builtin_amdgcn_wmma_f32_16x16x32_f16(
            false, af[i].v, false, bf[j].v, (short)0, acc[i][j].v, false, false);

    __syncthreads();
  }

  // fused epilogue: distance -> log_alpha into padded buffer
  float* laB = la + (size_t)b * LA_ROWS * LA_COLS;
  for (int j = 0; j < 4; ++j) {
    int col = m0 + wc * 64 + j * 16 + aM;
    float yv = yy[(size_t)b * N_PTS + col];
    for (int i = 0; i < 2; ++i) {
      for (int r = 0; r < 8; ++r) {
        int row = n0 + wr * 32 + i * 16 + ((lane < 16) ? r : r + 8);  // ISA C/D layout
        float xv = xx[(size_t)b * N_PTS + row];
        float dist = (xv + yv - 2.0f * acc[i][j].f[r]) * 0.0625f;  // / sqrt(256)
        laB[(size_t)row * LA_COLS + col] = 0.2f - dist;             // BETA*(ALPHA - d)
      }
    }
  }
}

// ---------------------------------------------------------------------------
// Sinkhorn row pass: rows 0..N-1, logsumexp over cols 0..N (incl. slack col)
__global__ __launch_bounds__(256) void row_lse_kernel(float* __restrict__ la) {
  int b = blockIdx.y, n = blockIdx.x, t = threadIdx.x;
  float* row = la + ((size_t)b * LA_ROWS + n) * LA_COLS;
  __shared__ float red[256];
  float mx = -1e30f;
  for (int m = t; m < LA_ROWS; m += 256) mx = fmaxf(mx, row[m]);
  red[t] = mx; __syncthreads();
  for (int s = 128; s > 0; s >>= 1) { if (t < s) red[t] = fmaxf(red[t], red[t + s]); __syncthreads(); }
  float bmax = red[0]; __syncthreads();
  float s = 0.0f;
  for (int m = t; m < LA_ROWS; m += 256) s += __expf(row[m] - bmax);
  red[t] = s; __syncthreads();
  for (int st = 128; st > 0; st >>= 1) { if (t < st) red[t] += red[t + st]; __syncthreads(); }
  float lse = bmax + __logf(red[0]);
  for (int m = t; m < LA_ROWS; m += 256) row[m] -= lse;
}

// Sinkhorn col pass: cols 0..N-1, logsumexp over rows 0..N (incl. slack row)
// 32 columns / block, online max+sum -> single read pass + subtract pass.
__global__ __launch_bounds__(256) void col_lse_kernel(float* __restrict__ la) {
  int b  = blockIdx.y;
  int ml = threadIdx.x & 31;
  int rg = threadIdx.x >> 5;            // 8 row groups
  int m  = blockIdx.x * 32 + ml;
  float* base = la + (size_t)b * LA_ROWS * LA_COLS + m;
  float mx = -1e30f, s = 0.0f;
  for (int n = rg; n < LA_ROWS; n += 8) {
    float x = base[(size_t)n * LA_COLS];
    if (x > mx) { s = s * __expf(mx - x) + 1.0f; mx = x; }
    else        { s += __expf(x - mx); }
  }
  __shared__ float smx[8][32], ssm[8][32];
  smx[rg][ml] = mx; ssm[rg][ml] = s;
  __syncthreads();
  float M = smx[0][ml], S = ssm[0][ml];
  for (int g = 1; g < 8; ++g) {
    float gm = smx[g][ml], gs = ssm[g][ml];
    if (gm > M) { S = S * __expf(M - gm) + gs; M = gm; }
    else        { S += gs * __expf(gm - M); }
  }
  float lse = M + __logf(S);
  for (int n = rg; n < LA_ROWS; n += 8) base[(size_t)n * LA_COLS] -= lse;
}

// ---------------------------------------------------------------------------
// exp interior -> output scores [B,N,N], plus row sums
__global__ __launch_bounds__(256) void exp_rows_kernel(
    const float* __restrict__ la, float* __restrict__ scores,
    float* __restrict__ rowsum) {
  int b = blockIdx.y, n = blockIdx.x, t = threadIdx.x;
  const float* row = la + ((size_t)b * LA_ROWS + n) * LA_COLS;
  float* srow = scores + ((size_t)b * N_PTS + n) * N_PTS;
  __shared__ float red[256];
  float s = 0.0f;
  for (int m = t; m < N_PTS; m += 256) { float e = __expf(row[m]); srow[m] = e; s += e; }
  red[t] = s; __syncthreads();
  for (int st = 128; st > 0; st >>= 1) { if (t < st) red[t] += red[t + st]; __syncthreads(); }
  if (t == 0) rowsum[(size_t)b * N_PTS + n] = red[0];
}

__device__ __forceinline__ float block_reduce_sum(float v, float* red) {
  int t = threadIdx.x;
  red[t] = v; __syncthreads();
  for (int s = 128; s > 0; s >>= 1) { if (t < s) red[t] += red[t + s]; __syncthreads(); }
  float r = red[0]; __syncthreads();
  return r;
}

// soft correspondences: corr[b,d,n] = sum_m tgt[b,d,m]*scores[b,n,m] / (rowsum+eps)
__global__ __launch_bounds__(256) void corr_kernel(
    const float* __restrict__ scores, const float* __restrict__ tgt,
    const float* __restrict__ rowsum, float* __restrict__ corr) {
  int b = blockIdx.y, n = blockIdx.x, t = threadIdx.x;
  const float* srow = scores + ((size_t)b * N_PTS + n) * N_PTS;
  const float* tg = tgt + (size_t)b * 3 * N_PTS;
  float a0 = 0, a1 = 0, a2 = 0;
  for (int m = t; m < N_PTS; m += 256) {
    float s = srow[m];
    a0 += s * tg[m];
    a1 += s * tg[N_PTS + m];
    a2 += s * tg[2 * N_PTS + m];
  }
  __shared__ float red[256];
  a0 = block_reduce_sum(a0, red);
  a1 = block_reduce_sum(a1, red);
  a2 = block_reduce_sum(a2, red);
  if (t == 0) {
    float inv = 1.0f / (rowsum[(size_t)b * N_PTS + n] + 1e-6f);
    float* cb = corr + (size_t)b * 3 * N_PTS;
    cb[n] = a0 * inv; cb[N_PTS + n] = a1 * inv; cb[2 * N_PTS + n] = a2 * inv;
  }
}

// ---------------------------------------------------------------------------
// per-batch: conf, centroids, 3x3 covariance, Jacobi SVD, Kabsch -> T[4,4]
__global__ __launch_bounds__(256) void svd_head_kernel(
    const float* __restrict__ src, const float* __restrict__ corr,
    const float* __restrict__ rowsum, float* __restrict__ outT) {
  int b = blockIdx.x, t = threadIdx.x;
  const float* sp = src  + (size_t)b * 3 * N_PTS;
  const float* cp = corr + (size_t)b * 3 * N_PTS;
  const float* rs = rowsum + (size_t)b * N_PTS;
  __shared__ float red[256];
  __shared__ float sh[8];

  float p = 0.0f;
  for (int n = t; n < N_PTS; n += 256) p += rs[n];
  p = block_reduce_sum(p, red);
  if (t == 0) sh[0] = p;
  __syncthreads();
  float inv_tot = 1.0f / (sh[0] + 1e-6f);

  float acc[6] = {0, 0, 0, 0, 0, 0};
  for (int n = t; n < N_PTS; n += 256) {
    float c = rs[n] * inv_tot;
    acc[0] += sp[n] * c; acc[1] += sp[N_PTS + n] * c; acc[2] += sp[2 * N_PTS + n] * c;
    acc[3] += cp[n] * c; acc[4] += cp[N_PTS + n] * c; acc[5] += cp[2 * N_PTS + n] * c;
  }
  for (int k = 0; k < 6; ++k) {
    float r = block_reduce_sum(acc[k], red);
    if (t == 0) sh[1 + k] = r;
  }
  __syncthreads();
  float cs0 = sh[1], cs1 = sh[2], cs2 = sh[3];
  float cc0 = sh[4], cc1 = sh[5], cc2 = sh[6];

  float h[9] = {0, 0, 0, 0, 0, 0, 0, 0, 0};
  for (int n = t; n < N_PTS; n += 256) {
    float c = rs[n] * inv_tot;
    float x0 = (sp[n] - cs0) * c, x1 = (sp[N_PTS + n] - cs1) * c, x2 = (sp[2 * N_PTS + n] - cs2) * c;
    float y0 = cp[n] - cc0, y1 = cp[N_PTS + n] - cc1, y2 = cp[2 * N_PTS + n] - cc2;
    h[0] += x0 * y0; h[1] += x0 * y1; h[2] += x0 * y2;
    h[3] += x1 * y0; h[4] += x1 * y1; h[5] += x1 * y2;
    h[6] += x2 * y0; h[7] += x2 * y1; h[8] += x2 * y2;
  }
  for (int k = 0; k < 9; ++k) h[k] = block_reduce_sum(h[k], red);

  if (t == 0) {
    float H[3][3] = {{h[0], h[1], h[2]}, {h[3], h[4], h[5]}, {h[6], h[7], h[8]}};
    float A[3][3], V[3][3] = {{1, 0, 0}, {0, 1, 0}, {0, 0, 1}};
    for (int i = 0; i < 3; ++i)
      for (int j = 0; j < 3; ++j)
        A[i][j] = H[0][i] * H[0][j] + H[1][i] * H[1][j] + H[2][i] * H[2][j];
    const int PP[3] = {0, 0, 1}, QQ[3] = {1, 2, 2};
    for (int sweep = 0; sweep < 12; ++sweep)
      for (int pi = 0; pi < 3; ++pi) {
        int p2 = PP[pi], q = QQ[pi];
        float apq = A[p2][q];
        if (fabsf(apq) < 1e-20f) continue;
        float theta = (A[q][q] - A[p2][p2]) / (2.0f * apq);
        float tt = 1.0f / (fabsf(theta) + sqrtf(theta * theta + 1.0f));
        if (theta < 0.0f) tt = -tt;
        float c = 1.0f / sqrtf(tt * tt + 1.0f), s2 = tt * c;
        for (int k = 0; k < 3; ++k) {
          float akp = A[k][p2], akq = A[k][q];
          A[k][p2] = c * akp - s2 * akq; A[k][q] = s2 * akp + c * akq;
        }
        for (int k = 0; k < 3; ++k) {
          float apk = A[p2][k], aqk = A[q][k];
          A[p2][k] = c * apk - s2 * aqk; A[q][k] = s2 * apk + c * aqk;
        }
        for (int k = 0; k < 3; ++k) {
          float vkp = V[k][p2], vkq = V[k][q];
          V[k][p2] = c * vkp - s2 * vkq; V[k][q] = s2 * vkp + c * vkq;
        }
      }
    // sort eigenpairs descending
    float lam[3] = {A[0][0], A[1][1], A[2][2]};
    int idx[3] = {0, 1, 2};
    for (int i = 0; i < 2; ++i)
      for (int j = i + 1; j < 3; ++j)
        if (lam[idx[j]] > lam[idx[i]]) { int tmp = idx[i]; idx[i] = idx[j]; idx[j] = tmp; }
    float Vs[3][3], U[3][3];
    for (int k = 0; k < 3; ++k)
      for (int i = 0; i < 3; ++i) Vs[i][k] = V[i][idx[k]];
    for (int k = 0; k < 3; ++k) {
      float ss = sqrtf(fmaxf(lam[idx[k]], 0.0f));
      float inv = 1.0f / (ss + 1e-12f);
      for (int i = 0; i < 3; ++i)
        U[i][k] = (H[i][0] * Vs[0][k] + H[i][1] * Vs[1][k] + H[i][2] * Vs[2][k]) * inv;
    }
    float R[3][3];
    for (int i = 0; i < 3; ++i)
      for (int j = 0; j < 3; ++j)
        R[i][j] = Vs[i][0] * U[j][0] + Vs[i][1] * U[j][1] + Vs[i][2] * U[j][2];
    float det = R[0][0] * (R[1][1] * R[2][2] - R[1][2] * R[2][1])
              - R[0][1] * (R[1][0] * R[2][2] - R[1][2] * R[2][0])
              + R[0][2] * (R[1][0] * R[2][1] - R[1][1] * R[2][0]);
    if (det <= 0.0f) {
      for (int i = 0; i < 3; ++i) Vs[i][2] = -Vs[i][2];
      for (int i = 0; i < 3; ++i)
        for (int j = 0; j < 3; ++j)
          R[i][j] = Vs[i][0] * U[j][0] + Vs[i][1] * U[j][1] + Vs[i][2] * U[j][2];
    }
    float cs[3] = {cs0, cs1, cs2}, cc[3] = {cc0, cc1, cc2};
    float* T = outT + (size_t)b * 16;
    for (int i = 0; i < 3; ++i) {
      float tv = -(R[i][0] * cs[0] + R[i][1] * cs[1] + R[i][2] * cs[2]) + cc[i];
      T[i * 4 + 0] = R[i][0]; T[i * 4 + 1] = R[i][1]; T[i * 4 + 2] = R[i][2]; T[i * 4 + 3] = tv;
    }
    T[12] = 0.0f; T[13] = 0.0f; T[14] = 0.0f; T[15] = 1.0f;
  }
}

// ---------------------------------------------------------------------------
extern "C" void kernel_launch(void* const* d_in, const int* in_sizes, int n_in,
                              void* d_out, int out_size, void* d_ws, size_t ws_size,
                              hipStream_t stream) {
  const float* srcE = (const float*)d_in[0];   // [B,256,2048]
  const float* tgtE = (const float*)d_in[1];   // [B,256,2048]
  const float* src  = (const float*)d_in[2];   // [B,3,2048]
  const float* tgt  = (const float*)d_in[3];   // [B,3,2048]

  float* outT = (float*)d_out;                 // [B,4,4]
  float* outS = outT + B_SZ * 16;              // [B,2048,2048]

  const size_t LA_TOTAL = (size_t)B_SZ * LA_ROWS * LA_COLS;
  const size_t BND = (size_t)B_SZ * N_PTS * D_DIM;
  const size_t BN  = (size_t)B_SZ * N_PTS;

  float* la     = (float*)d_ws;
  f16*   srcH   = (f16*)(la + LA_TOTAL);
  f16*   tgtH   = srcH + BND;
  float* xx     = (float*)(tgtH + BND);
  float* yy     = xx + BN;
  float* rowsum = yy + BN;
  float* corr   = rowsum + BN;                 // [B,3,2048]

  // zero padded log_alpha (slack row/col + stride padding)
  zero_kernel<<<4096, 256, 0, stream>>>(la, (long long)LA_TOTAL);

  // f16 transposed embeddings + fp32 norms
  dim3 tb(32, 8);
  transpose_f16_kernel<<<dim3(64, 8, B_SZ), tb, 0, stream>>>(srcE, srcH);
  transpose_f16_kernel<<<dim3(64, 8, B_SZ), tb, 0, stream>>>(tgtE, tgtH);
  norm_kernel<<<dim3(8, B_SZ), 256, 0, stream>>>(srcE, xx);
  norm_kernel<<<dim3(8, B_SZ), 256, 0, stream>>>(tgtE, yy);

  // WMMA distance GEMM -> log_alpha
  gemm_scores_kernel<<<dim3(16, 16, B_SZ), 256, 0, stream>>>(srcH, tgtH, xx, yy, la);

  // Sinkhorn (L2-resident: 135 MB < 192 MB L2)
  for (int it = 0; it < 5; ++it) {
    row_lse_kernel<<<dim3(N_PTS, B_SZ), 256, 0, stream>>>(la);
    col_lse_kernel<<<dim3(64, B_SZ), 256, 0, stream>>>(la);
  }

  exp_rows_kernel<<<dim3(N_PTS, B_SZ), 256, 0, stream>>>(la, outS, rowsum);
  corr_kernel<<<dim3(N_PTS, B_SZ), 256, 0, stream>>>(outS, tgt, rowsum, corr);
  svd_head_kernel<<<B_SZ, 256, 0, stream>>>(src, corr, rowsum, outT);
}
